// LlamaAttention_62801011802987
// MI455X (gfx1250) — matrix operand order repaired and checked by reading, not compiled, not run on hardware
//
#include <hip/hip_runtime.h>
#include <hip/hip_bf16.h>
#include <math.h>

typedef __bf16 bf16_t;
typedef __attribute__((ext_vector_type(16))) __bf16 v16bf;
typedef __attribute__((ext_vector_type(8)))  float  v8f;

#define EPSQ 1e-5f

union FragBF {
    uint4 q[2];
    v16bf v;
};

__device__ __forceinline__ void async_b128_to_lds(const void* gptr, void* lptr) {
    uint32_t l = (uint32_t)(uintptr_t)lptr;          // LDS byte address (low 32 bits)
    uint64_t g = (uint64_t)(uintptr_t)gptr;
    asm volatile("global_load_async_to_lds_b128 %0, %1, off"
                 :: "v"(l), "v"(g) : "memory");
}

__device__ __forceinline__ void wait_async0() {
    asm volatile("s_wait_asynccnt 0" ::: "memory");
}

// ---------------- sum(|w|) reduction ----------------
__global__ void k_abssum(const float* __restrict__ w, float* __restrict__ sum, int n) {
    __shared__ float red[256];
    float acc = 0.f;
    for (int i = blockIdx.x * blockDim.x + threadIdx.x; i < n; i += gridDim.x * blockDim.x)
        acc += fabsf(w[i]);
    red[threadIdx.x] = acc;
    __syncthreads();
    for (int s = 128; s > 0; s >>= 1) {
        if (threadIdx.x < s) red[threadIdx.x] += red[threadIdx.x + s];
        __syncthreads();
    }
    if (threadIdx.x == 0) atomicAdd(sum, red[0]);
}

// ---------------- ternarize weight to bf16 {-1,0,1} ----------------
__global__ void k_quant(const float* __restrict__ w, const float* __restrict__ sum,
                        bf16_t* __restrict__ qw, int n, float inv_cnt) {
    int i = blockIdx.x * blockDim.x + threadIdx.x;
    if (i >= n) return;
    float scale = sum[0] * inv_cnt;          // mean(|w|)
    float t = w[i] / (scale + EPSQ);
    t = fminf(fmaxf(t, -1.f), 1.f);
    qw[i] = (bf16_t)rintf(t);                // round-to-nearest-even like jnp.round
}

// ---------------- fp32 -> bf16 ----------------
__global__ void k_f32_to_bf16(const float* __restrict__ in, bf16_t* __restrict__ out, long n) {
    long i = (long)blockIdx.x * blockDim.x + threadIdx.x;
    if (i < n) out[i] = (bf16_t)in[i];
}

// ---------------- RoPE on q,k (fp32 in) -> bf16 out ----------------
// layout: (B=2, S=4096, H=2048) viewed as (..., 16 heads, 128); pairs (d, d+64)
__global__ void k_rope(const float* __restrict__ q32, const float* __restrict__ k32,
                       const float* __restrict__ cosp, const float* __restrict__ sinp,
                       bf16_t* __restrict__ qb, bf16_t* __restrict__ kb) {
    long i = (long)blockIdx.x * blockDim.x + threadIdx.x;  // < 2*4096*2048
    int d = (int)(i & 127);
    int s = (int)((i >> 11) & 4095);
    float c  = cosp[s * 128 + d];
    float sn = sinp[s * 128 + d];
    float q0 = q32[i], k0 = k32[i];
    float qr, kr;
    if (d < 64) {
        qr = q0 * c - q32[i + 64] * sn;
        kr = k0 * c - k32[i + 64] * sn;
    } else {
        qr = q0 * c + q32[i - 64] * sn;
        kr = k0 * c + k32[i - 64] * sn;
    }
    qb[i] = (bf16_t)qr;
    kb[i] = (bf16_t)kr;
}

// ---------------- V (B,S,H) fp32 -> V^T (B,H,S) bf16 ----------------
__global__ void k_transpose_v(const float* __restrict__ v32, bf16_t* __restrict__ vT) {
    __shared__ float tile[32][33];
    int b = blockIdx.z;
    int s0 = blockIdx.x * 32, h0 = blockIdx.y * 32;
    int tx = threadIdx.x & 31, ty = threadIdx.x >> 5;   // 32 x 8
    const float* src = v32 + (long)b * 4096 * 2048;
    bf16_t* dst = vT + (long)b * 2048 * 4096;
    #pragma unroll
    for (int i = 0; i < 4; i++)
        tile[ty + i * 8][tx] = src[(long)(s0 + ty + i * 8) * 2048 + h0 + tx];
    __syncthreads();
    #pragma unroll
    for (int i = 0; i < 4; i++)
        dst[(long)(h0 + ty + i * 8) * 4096 + s0 + tx] = (bf16_t)tile[tx][ty + i * 8];
}

// ---------------- WMMA GEMM: out[M,N] = scale * X[M,K] @ W[N,K]^T ----------------
// ldX = ldW = K, ldOut = N. Block tile 128x128x32, 8 waves, wave tile 64x32.
// Global->LDS staging uses CDNA5 async loads (ASYNCcnt), double-buffered.
__global__ __launch_bounds__(256) void k_gemm_bf16(
    const bf16_t* __restrict__ X, const bf16_t* __restrict__ W, float* __restrict__ out,
    int N, int K, long sx, long sw, long so,
    const float* __restrict__ sumabs, float inv_cnt, float cscale)
{
    __shared__ __align__(16) bf16_t As[2][128][32];
    __shared__ __align__(16) bf16_t Bs[2][128][32];
    const int tid = threadIdx.x;
    const int lane = tid & 31, wid = tid >> 5;
    const int wm = (wid >> 2) * 64;   // 2 waves along M
    const int wn = (wid & 3) * 32;    // 4 waves along N
    const long rowX0 = (long)blockIdx.x * 128;
    const long colW0 = (long)blockIdx.y * 128;
    X   += (long)blockIdx.z * sx;
    W   += (long)blockIdx.z * sw;
    out += (long)blockIdx.z * so;

    v8f acc[4][2] = {};

    auto loadTilesAsync = [&](int buf, int k0) {
        #pragma unroll
        for (int i = 0; i < 2; i++) {
            int c  = tid + i * 256;      // 0..511
            int r  = c >> 2;             // row 0..127
            int cc = (c & 3) * 8;        // k element 0,8,16,24 (16B chunks)
            async_b128_to_lds(X + (rowX0 + r) * (long)K + k0 + cc, &As[buf][r][cc]);
            async_b128_to_lds(W + (colW0 + r) * (long)K + k0 + cc, &Bs[buf][r][cc]);
        }
    };

    loadTilesAsync(0, 0);
    wait_async0();
    __syncthreads();
    const int nk = K / 32;
    for (int kt = 0; kt < nk; ++kt) {
        int buf = kt & 1;
        if (kt + 1 < nk) loadTilesAsync(buf ^ 1, (kt + 1) * 32);

        FragBF a[4], bb[2];
        // A fragment (16x32 bf16): lanes 0-15 hold K 0..7 & 16..23, lanes 16-31 hold K 8..15 & 24..31
        int kq = (lane >> 4) * 8;
        #pragma unroll
        for (int mt = 0; mt < 4; mt++) {
            int r = wm + mt * 16 + (lane & 15);
            a[mt].q[0] = *(const uint4*)&As[buf][r][kq];
            a[mt].q[1] = *(const uint4*)&As[buf][r][kq + 16];
        }
        // B fragment (32x16 = W^T tile): lanes 0-15 hold K 0..7 & 16..23, lanes 16-31 hold K 8..15 & 24..31 of column n=lane&15
        int kh = (lane >> 4) * 8;
        #pragma unroll
        for (int nt = 0; nt < 2; nt++) {
            int r = wn + nt * 16 + (lane & 15);
            bb[nt].q[0] = *(const uint4*)&Bs[buf][r][kh];
            bb[nt].q[1] = *(const uint4*)&Bs[buf][r][kh + 16];
        }
        #pragma unroll
        for (int mt = 0; mt < 4; mt++)
            #pragma unroll
            for (int nt = 0; nt < 2; nt++)
                acc[mt][nt] = __builtin_amdgcn_wmma_f32_16x16x32_bf16(
                    false, a[mt].v, false, bb[nt].v, (short)0, acc[mt][nt], false, false);

        wait_async0();        // async writes into buf^1 complete
        __syncthreads();      // all waves done reading buf / writing buf^1
    }

    float sc = cscale;
    if (sumabs) sc *= sumabs[0] * inv_cnt;   // BitLinear dequant scale
    #pragma unroll
    for (int mt = 0; mt < 4; mt++) {
        #pragma unroll
        for (int nt = 0; nt < 2; nt++) {
            int n = wn + nt * 16 + (lane & 15);
            #pragma unroll
            for (int v = 0; v < 8; v++) {
                int m = wm + mt * 16 + (lane >> 4) * 8 + v;   // C layout: VGPR v -> M=v(+8)
                out[(rowX0 + m) * (long)N + colW0 + n] = acc[mt][nt][v] * sc;
            }
        }
    }
}

// ---------------- row softmax: fp32 scores + mask -> bf16 probs ----------------
__global__ void k_softmax(const float* __restrict__ scores, const float* __restrict__ mask,
                          bf16_t* __restrict__ P) {
    __shared__ float red[256];
    const int row = blockIdx.x;          // 0..8191 (b*4096+q)
    const int q = row & 4095;
    const float* srow = scores + (long)row * 4096;
    const float* mrow = mask + (long)q * 4096;   // mask batch dim is 1
    float m = -INFINITY;
    float vals[16];
    #pragma unroll
    for (int i = 0; i < 16; i++) {
        int k = threadIdx.x + i * 256;
        float v = srow[k] + mrow[k];
        vals[i] = v;
        m = fmaxf(m, v);
    }
    red[threadIdx.x] = m; __syncthreads();
    for (int s = 128; s > 0; s >>= 1) {
        if (threadIdx.x < s) red[threadIdx.x] = fmaxf(red[threadIdx.x], red[threadIdx.x + s]);
        __syncthreads();
    }
    m = red[0]; __syncthreads();
    float sum = 0.f;
    #pragma unroll
    for (int i = 0; i < 16; i++) { vals[i] = __expf(vals[i] - m); sum += vals[i]; }
    red[threadIdx.x] = sum; __syncthreads();
    for (int s = 128; s > 0; s >>= 1) {
        if (threadIdx.x < s) red[threadIdx.x] += red[threadIdx.x + s];
        __syncthreads();
    }
    float rl = 1.f / red[0];
    #pragma unroll
    for (int i = 0; i < 16; i++)
        P[(long)row * 4096 + threadIdx.x + i * 256] = (bf16_t)(vals[i] * rl);
}

extern "C" void kernel_launch(void* const* d_in, const int* in_sizes, int n_in,
                              void* d_out, int out_size, void* d_ws, size_t ws_size,
                              hipStream_t stream) {
    const float* hs   = (const float*)d_in[0];
    const float* mask = (const float*)d_in[1];
    const float* cosp = (const float*)d_in[2];
    const float* sinp = (const float*)d_in[3];
    const float* wsrc[4] = { (const float*)d_in[4], (const float*)d_in[5],
                             (const float*)d_in[6], (const float*)d_in[7] };

    const long B = 2, S = 4096, H = 2048;
    const long MB = 1024 * 1024;
    char* ws = (char*)d_ws;

    // workspace layout (aliased where lifetimes allow)
    float* sums = (float*)ws;                      // 4 floats
    size_t off = 256;
    bf16_t* qw[4];
    for (int i = 0; i < 4; i++) { qw[i] = (bf16_t*)(ws + off); off += 8 * MB; }
    bf16_t* xbf = (bf16_t*)(ws + off); size_t off_xbf = off; off += 32 * MB;
    float*  q32 = (float*)(ws + off);  size_t off_q32 = off; off += 64 * MB;
    float*  k32 = (float*)(ws + off);                        off += 64 * MB;
    float*  v32 = (float*)(ws + off);  size_t off_v32 = off; off += 64 * MB;
    bf16_t* qbf = (bf16_t*)(ws + off); size_t off_qbf = off; off += 32 * MB;
    bf16_t* kbf = (bf16_t*)(ws + off);                       off += 32 * MB;
    bf16_t* vT  = (bf16_t*)(ws + off);                       off += 32 * MB;
    float*  scores = (float*)(ws + off_q32);   // 128MB, reuses q32+k32 (dead after rope)
    bf16_t* probs  = (bf16_t*)(ws + off_v32);  // 64MB, reuses v32 (dead after transpose)
    float*  attn32 = (float*)(ws + off_qbf);   // 64MB, reuses qbf+kbf (dead after scores)
    bf16_t* attnbf = (bf16_t*)(ws + off_xbf);  // 32MB, reuses xbf

    hipMemsetAsync(sums, 0, 4 * sizeof(float), stream);

    const int WN = 2048 * 2048;
    const float inv_cnt = 1.0f / (float)WN;
    for (int i = 0; i < 4; i++)
        k_abssum<<<1024, 256, 0, stream>>>(wsrc[i], sums + i, WN);
    for (int i = 0; i < 4; i++)
        k_quant<<<WN / 256, 256, 0, stream>>>(wsrc[i], sums + i, qw[i], WN, inv_cnt);

    const long NH = B * S * H;  // 16M
    k_f32_to_bf16<<<(int)(NH / 256), 256, 0, stream>>>(hs, xbf, NH);

    dim3 gProj(8192 / 128, 2048 / 128, 1);
    k_gemm_bf16<<<gProj, 256, 0, stream>>>(xbf, qw[0], q32, 2048, 2048, 0, 0, 0, sums + 0, inv_cnt, 1.0f);
    k_gemm_bf16<<<gProj, 256, 0, stream>>>(xbf, qw[1], k32, 2048, 2048, 0, 0, 0, sums + 1, inv_cnt, 1.0f);
    k_gemm_bf16<<<gProj, 256, 0, stream>>>(xbf, qw[2], v32, 2048, 2048, 0, 0, 0, sums + 2, inv_cnt, 1.0f);

    k_rope<<<(int)(NH / 256), 256, 0, stream>>>(q32, k32, cosp, sinp, qbf, kbf);
    dim3 gT(4096 / 32, 2048 / 32, 2);
    k_transpose_v<<<gT, 256, 0, stream>>>(v32, vT);

    // scores = (Q @ K^T) / sqrt(128), per batch
    dim3 gScore(4096 / 128, 4096 / 128, 2);
    k_gemm_bf16<<<gScore, 256, 0, stream>>>(qbf, kbf, scores, 4096, 2048,
        S * H, S * H, S * S, nullptr, 0.f, 0.08838834764831845f);

    k_softmax<<<(int)(B * S), 256, 0, stream>>>(scores, mask, probs);

    // attn = P @ V  (V^T rows are the "W" matrix: W[h][k]=V[k][h])
    dim3 gPV(4096 / 128, 2048 / 128, 2);
    k_gemm_bf16<<<gPV, 256, 0, stream>>>(probs, vT, attn32, 2048, 4096,
        S * S, H * S, S * H, nullptr, 0.f, 1.0f);

    k_f32_to_bf16<<<(int)(NH / 256), 256, 0, stream>>>(attn32, attnbf, NH);
    k_gemm_bf16<<<gProj, 256, 0, stream>>>(attnbf, qw[3], (float*)d_out, 2048, 2048,
        0, 0, 0, sums + 3, inv_cnt, 1.0f);
}
